// HMMPoisson_34033320853827
// MI455X (gfx1250) — matrix operand order repaired and checked by reading, not compiled, not run on hardware
//
#include <hip/hip_runtime.h>
#include <hip/hip_bf16.h>

typedef __attribute__((ext_vector_type(16))) _Float16 v16h;
typedef __attribute__((ext_vector_type(8)))  float    v8f;

#define T_LEN   131072
#define NST     64
#define MDIM    8
#define NCHUNK  1024
#define CLEN    128          // steps per chunk over the T-1 = 131071 scan steps
#define NGROUP  64
#define GSIZE   16
#define SC_UP    10.0f       // operands scaled by 2^10 (log2 domain: +10)
#define SC_DOWN  20.0f       // product of two scaled operands: -20 in log2
#define LOG2E   1.44269504088896341f
#define LN2     0.69314718055994531f
#define TINY    1e-30f       // additive floor: keeps log2 input strictly positive

union HF  { v16h v; uint32_t u[8]; };
union CF  { v8f  v; float    f[8]; };
union H2U { _Float16 h[2]; uint32_t u; };

// raw hardware transcendentals: v_exp_f32 (exp2), v_log_f32 (log2)
static __device__ __forceinline__ float fexp2(float x) { return __builtin_amdgcn_exp2f(x); }
static __device__ __forceinline__ float flog2(float x) { return __builtin_amdgcn_logf(x); }

// ---- WMMA fragment loaders (layouts per CDNA5 ISA 7.12.2, wave32) ----------
// A-matrix 16x32 f16: lane half h=lane>>4, row m=lane&15(+16*tileRow);
//   v<4: K = 2v + 8h (+32*kchunk);  v>=4: K = 16 + 2(v-4) + 8h (+32*kchunk)
static __device__ __forceinline__ v16h load_a_frag(const _Float16* base, int strideH,
                                                   int tileRow, int kchunk, int lane) {
  const int h = lane >> 4;
  const int m = (lane & 15) + 16 * tileRow;
  HF a;
#pragma unroll
  for (int v = 0; v < 8; ++v) {
    int kb = (v < 4) ? (2 * v + 8 * h) : (16 + 2 * (v - 4) + 8 * h);
    kb += 32 * kchunk;
    a.u[v] = *(const uint32_t*)(base + m * strideH + kb);
  }
  return a.v;
}
// B-matrix 32x16 f16 loaded from a TRANSPOSED store PT[n][k] (so K pairs are
// contiguous): same K order as A: v<4: K = 2v + 8h; v>=4: K = 16 + 2(v-4) + 8h.
static __device__ __forceinline__ v16h load_b_frag(const _Float16* basePT, int strideH,
                                                   int tileCol, int kchunk, int lane) {
  const int h = lane >> 4;
  const int n = (lane & 15) + 16 * tileCol;
  HF b;
#pragma unroll
  for (int v = 0; v < 8; ++v) {
    const int k = ((v < 4) ? (2 * v + 8 * h) : (16 + 2 * (v - 4) + 8 * h)) + 32 * kchunk;
    b.u[v] = *(const uint32_t*)(basePT + n * strideH + k);
  }
  return b.v;
}

// ---- K0: constants (everything pre-converted to the log2 domain) ----------
__global__ __launch_bounds__(256) void hmm_prep(const float* __restrict__ lambdas,
                                                const float* __restrict__ logA,
                                                const float* __restrict__ priors,
                                                _Float16* __restrict__ Tlin,
                                                float* __restrict__ loglam2,
                                                float* __restrict__ lamsum2,
                                                float* __restrict__ logfact2,
                                                float* __restrict__ priors2) {
  const int tid = threadIdx.x;
  for (int idx = tid; idx < NST * NST; idx += 256)
    Tlin[idx] = (_Float16)fexp2(logA[idx] * LOG2E + SC_UP);   // exp(A) * 2^10
  for (int idx = tid; idx < NST * MDIM; idx += 256)
    loglam2[idx] = log2f(lambdas[idx]);
  if (tid < NST) {
    float s = 0.0f;
#pragma unroll
    for (int m = 0; m < MDIM; ++m) s += lambdas[tid * MDIM + m];
    lamsum2[tid] = s * LOG2E;
    priors2[tid] = priors[tid] * LOG2E;
  }
  if (tid < 32) logfact2[tid] = lgammaf((float)tid + 1.0f) * LOG2E;
}

// ---- K1: per-chunk scan -> 64x64 chunk operator matrix (log2 domain) ------
__global__ __launch_bounds__(256) void hmm_chunk_scan(const int* __restrict__ x,
                                                      const _Float16* __restrict__ Tlin,
                                                      const float* __restrict__ loglamG,
                                                      const float* __restrict__ lamsumG,
                                                      const float* __restrict__ logfactG,
                                                      float* __restrict__ Mc) {
  __shared__ __align__(16) float    Mlog[NST * 65];
  __shared__ __align__(16) _Float16 PT[NST * 66];
  __shared__ float emS[NST];
  __shared__ float colmax[NST];
  __shared__ float pmax[NST][4];
  __shared__ float loglamS[NST * MDIM];
  __shared__ float lamsumS[NST];
  __shared__ float logfactS[32];

  const int tid  = threadIdx.x;
  const int lane = tid & 31;
  const int w    = tid >> 5;          // wave 0..7
  const int tileRow  = w >> 1;        // 0..3
  const int tileCol0 = (2 * w) & 3;   // 0 or 2
  const int c = blockIdx.x;
  const int jq = tid >> 2;            // column 0..63
  const int q  = tid & 3;             // quarter 0..3

  if (tid < NST * MDIM) loglamS[tid] = loglamG[tid];
  if (tid < NST)        lamsumS[tid] = lamsumG[tid];
  if (tid < 32)         logfactS[tid] = logfactG[tid];
  for (int r = 0; r < 16; ++r) {      // log-identity
    const int idx = tid + 256 * r;
    const int i = idx >> 6, j = idx & 63;
    Mlog[i * 65 + j] = (i == j) ? 0.0f : -1e30f;
  }
  const v16h a0 = load_a_frag(Tlin, 64, tileRow, 0, lane);   // constant A frags
  const v16h a1 = load_a_frag(Tlin, 64, tileRow, 1, lane);
  __syncthreads();

  const int tStart = 1 + c * CLEN;
  int tEnd = tStart + CLEN;
  if (tEnd > T_LEN) tEnd = T_LEN;

  for (int t = tStart; t < tEnd; ++t) {
    if (tid < NST) {                            // emission (log2 units)
      float e = -lamsumS[tid];
#pragma unroll
      for (int m = 0; m < MDIM; ++m) {
        const int xv = x[t * MDIM + m];
        e += (float)xv * loglamS[tid * MDIM + m] - logfactS[xv & 31];
      }
      emS[tid] = e;
    }
    float cache[16];                            // column slice + partial max
    float pm = -1e30f;
#pragma unroll
    for (int kk = 0; kk < 16; ++kk) {
      const float v = Mlog[(16 * q + kk) * 65 + jq];
      cache[kk] = v;
      pm = fmaxf(pm, v);
    }
    pmax[jq][q] = pm;
    __syncthreads();
    if (tid < NST)
      colmax[tid] = fmaxf(fmaxf(pmax[tid][0], pmax[tid][1]),
                          fmaxf(pmax[tid][2], pmax[tid][3]));
    __syncthreads();
    {                                 // P^T[j][k] = 2^(M[k][j]-m_j+10), dword stores
      const float cm = colmax[jq] - SC_UP;
#pragma unroll
      for (int kk = 0; kk < 16; kk += 2) {
        H2U p;
        p.h[0] = (_Float16)fexp2(cache[kk]     - cm);
        p.h[1] = (_Float16)fexp2(cache[kk + 1] - cm);
        *(uint32_t*)&PT[jq * 66 + 16 * q + kk] = p.u;
      }
    }
    __syncthreads();
    v8f acc0 = {}; v8f acc1 = {};
    const v16h b00 = load_b_frag(PT, 66, tileCol0,     0, lane);
    const v16h b01 = load_b_frag(PT, 66, tileCol0,     1, lane);
    const v16h b10 = load_b_frag(PT, 66, tileCol0 + 1, 0, lane);
    const v16h b11 = load_b_frag(PT, 66, tileCol0 + 1, 1, lane);
    acc0 = __builtin_amdgcn_wmma_f32_16x16x32_f16(false, a0, false, b00, (short)0, acc0, false, false);
    acc0 = __builtin_amdgcn_wmma_f32_16x16x32_f16(false, a1, false, b01, (short)0, acc0, false, false);
    acc1 = __builtin_amdgcn_wmma_f32_16x16x32_f16(false, a0, false, b10, (short)0, acc1, false, false);
    acc1 = __builtin_amdgcn_wmma_f32_16x16x32_f16(false, a1, false, b11, (short)0, acc1, false, false);
    {                                           // back to log2 domain
      CF c0, c1; c0.v = acc0; c1.v = acc1;
      const int h = lane >> 4;
      const int j0 = 16 * tileCol0 + (lane & 15);
      const int j1 = j0 + 16;
      const float cm0 = colmax[j0] - SC_DOWN, cm1 = colmax[j1] - SC_DOWN;
#pragma unroll
      for (int r = 0; r < 8; ++r) {
        const int i = 16 * tileRow + 8 * h + r;
        const float ei = emS[i];
        Mlog[i * 65 + j0] = ei + cm0 + flog2(c0.f[r] + TINY);
        Mlog[i * 65 + j1] = ei + cm1 + flog2(c1.f[r] + TINY);
      }
    }
    __syncthreads();
  }
  float* dst = Mc + (size_t)c * 4096;
  for (int r = 0; r < 16; ++r) {
    const int idx = tid + 256 * r;
    dst[idx] = Mlog[(idx >> 6) * 65 + (idx & 63)];
  }
}

// ---- K2: fold 16 chunk matrices -> 1 (WMMA log2-matmul) -------------------
__global__ __launch_bounds__(256) void hmm_group_compose(const float* __restrict__ Mc,
                                                         float* __restrict__ Msuper) {
  __shared__ __align__(16) float    Alog[NST * 65];   // accumulated (older)
  __shared__ __align__(16) float    Blog[NST * 65];   // newer chunk
  __shared__ __align__(16) _Float16 PT[NST * 66];     // 2^(Alog - colmax + 10)
  __shared__ __align__(16) _Float16 QA[NST * 66];     // 2^(Blog - rowmax + 10)
  __shared__ float colmax[NST], rowmax[NST];
  __shared__ float pmaxC[NST][4], pmaxR[NST][4];

  const int tid  = threadIdx.x;
  const int lane = tid & 31;
  const int w    = tid >> 5;
  const int tileRow  = w >> 1;
  const int tileCol0 = (2 * w) & 3;
  const int rq = tid >> 2, q = tid & 3;
  const float* base = Mc + (size_t)blockIdx.x * GSIZE * 4096;

  for (int r = 0; r < 16; ++r) {
    const int idx = tid + 256 * r;
    Alog[(idx >> 6) * 65 + (idx & 63)] = base[idx];
  }
  __syncthreads();

  for (int s = 1; s < GSIZE; ++s) {
    for (int r = 0; r < 16; ++r) {
      const int idx = tid + 256 * r;
      Blog[(idx >> 6) * 65 + (idx & 63)] = base[(size_t)s * 4096 + idx];
    }
    __syncthreads();
    float cc[16], cr[16];
    float pmc = -1e30f, pmr = -1e30f;
#pragma unroll
    for (int kk = 0; kk < 16; ++kk) {
      const float va = Alog[(16 * q + kk) * 65 + rq];   // column rq of Alog
      cc[kk] = va; pmc = fmaxf(pmc, va);
      const float vb = Blog[rq * 65 + 16 * q + kk];     // row rq of Blog
      cr[kk] = vb; pmr = fmaxf(pmr, vb);
    }
    pmaxC[rq][q] = pmc; pmaxR[rq][q] = pmr;
    __syncthreads();
    if (tid < NST) {
      colmax[tid] = fmaxf(fmaxf(pmaxC[tid][0], pmaxC[tid][1]),
                          fmaxf(pmaxC[tid][2], pmaxC[tid][3]));
      rowmax[tid] = fmaxf(fmaxf(pmaxR[tid][0], pmaxR[tid][1]),
                          fmaxf(pmaxR[tid][2], pmaxR[tid][3]));
    }
    __syncthreads();
    {
      const float cm = colmax[rq] - SC_UP, rm = rowmax[rq] - SC_UP;
#pragma unroll
      for (int kk = 0; kk < 16; kk += 2) {
        H2U p, qq;
        p.h[0]  = (_Float16)fexp2(cc[kk]     - cm);
        p.h[1]  = (_Float16)fexp2(cc[kk + 1] - cm);
        qq.h[0] = (_Float16)fexp2(cr[kk]     - rm);
        qq.h[1] = (_Float16)fexp2(cr[kk + 1] - rm);
        *(uint32_t*)&PT[rq * 66 + 16 * q + kk] = p.u;
        *(uint32_t*)&QA[rq * 66 + 16 * q + kk] = qq.u;
      }
    }
    __syncthreads();
    const v16h a0  = load_a_frag(QA, 66, tileRow, 0, lane);
    const v16h a1  = load_a_frag(QA, 66, tileRow, 1, lane);
    const v16h b00 = load_b_frag(PT, 66, tileCol0,     0, lane);
    const v16h b01 = load_b_frag(PT, 66, tileCol0,     1, lane);
    const v16h b10 = load_b_frag(PT, 66, tileCol0 + 1, 0, lane);
    const v16h b11 = load_b_frag(PT, 66, tileCol0 + 1, 1, lane);
    v8f acc0 = {}; v8f acc1 = {};
    acc0 = __builtin_amdgcn_wmma_f32_16x16x32_f16(false, a0, false, b00, (short)0, acc0, false, false);
    acc0 = __builtin_amdgcn_wmma_f32_16x16x32_f16(false, a1, false, b01, (short)0, acc0, false, false);
    acc1 = __builtin_amdgcn_wmma_f32_16x16x32_f16(false, a0, false, b10, (short)0, acc1, false, false);
    acc1 = __builtin_amdgcn_wmma_f32_16x16x32_f16(false, a1, false, b11, (short)0, acc1, false, false);
    {
      CF c0, c1; c0.v = acc0; c1.v = acc1;
      const int h = lane >> 4;
      const int j0 = 16 * tileCol0 + (lane & 15);
      const int j1 = j0 + 16;
      const float cm0 = colmax[j0] - SC_DOWN, cm1 = colmax[j1] - SC_DOWN;
#pragma unroll
      for (int r = 0; r < 8; ++r) {
        const int i = 16 * tileRow + 8 * h + r;
        const float rm = rowmax[i];
        Alog[i * 65 + j0] = rm + cm0 + flog2(c0.f[r] + TINY);
        Alog[i * 65 + j1] = rm + cm1 + flog2(c1.f[r] + TINY);
      }
    }
    __syncthreads();
  }
  float* dst = Msuper + (size_t)blockIdx.x * 4096;
  for (int r = 0; r < 16; ++r) {
    const int idx = tid + 256 * r;
    dst[idx] = Alog[(idx >> 6) * 65 + (idx & 63)];
  }
}

// ---- K3: alpha0, apply 64 super-matrices, final logsumexp -----------------
__global__ __launch_bounds__(256) void hmm_finalize(const int* __restrict__ x,
                                                    const float* __restrict__ loglamG,
                                                    const float* __restrict__ lamsumG,
                                                    const float* __restrict__ logfactG,
                                                    const float* __restrict__ priors2,
                                                    const float* __restrict__ Msuper,
                                                    float* __restrict__ out) {
  __shared__ float alpha[NST];
  __shared__ float Ml[NST * 65];
  const int tid = threadIdx.x;
  if (tid < NST) {
    float e = priors2[tid] - lamsumG[tid];
#pragma unroll
    for (int m = 0; m < MDIM; ++m) {
      const int xv = x[m];                       // t = 0
      e += (float)xv * loglamG[tid * MDIM + m] - logfactG[xv & 31];
    }
    alpha[tid] = e;
  }
  __syncthreads();
  for (int g = 0; g < NGROUP; ++g) {
    const float* src = Msuper + (size_t)g * 4096;
    for (int r = 0; r < 16; ++r) {
      const int idx = tid + 256 * r;
      Ml[(idx >> 6) * 65 + (idx & 63)] = src[idx];
    }
    __syncthreads();
    float anew = 0.0f;
    if (tid < NST) {
      float m1 = -3.0e38f;
#pragma unroll 4
      for (int j = 0; j < NST; ++j) m1 = fmaxf(m1, Ml[tid * 65 + j] + alpha[j]);
      float s = 0.0f;
#pragma unroll 4
      for (int j = 0; j < NST; ++j) s += fexp2(Ml[tid * 65 + j] + alpha[j] - m1);
      anew = m1 + flog2(s);
    }
    __syncthreads();
    if (tid < NST) alpha[tid] = anew;
    __syncthreads();
  }
  if (tid == 0) {
    float m1 = -3.0e38f;
    for (int j = 0; j < NST; ++j) m1 = fmaxf(m1, alpha[j]);
    float s = 0.0f;
    for (int j = 0; j < NST; ++j) s += fexp2(alpha[j] - m1);
    out[0] = (m1 + flog2(s)) * LN2;              // back to natural log
  }
}

extern "C" void kernel_launch(void* const* d_in, const int* in_sizes, int n_in,
                              void* d_out, int out_size, void* d_ws, size_t ws_size,
                              hipStream_t stream) {
  (void)in_sizes; (void)n_in; (void)out_size; (void)ws_size;
  const int*   x       = (const int*)d_in[0];
  const float* lambdas = (const float*)d_in[1];
  const float* logA    = (const float*)d_in[2];
  const float* priors  = (const float*)d_in[3];
  float* out = (float*)d_out;

  char* ws = (char*)d_ws;
  float*    Mc       = (float*)ws;                                   // 16 MB
  float*    Msuper   = (float*)(ws + (size_t)16 * 1024 * 1024);      //  1 MB
  char*     p        = ws + (size_t)17 * 1024 * 1024;
  _Float16* Tlin     = (_Float16*)p;         p += NST * NST * sizeof(_Float16);
  float*    loglam2  = (float*)p;            p += NST * MDIM * sizeof(float);
  float*    lamsum2  = (float*)p;            p += NST * sizeof(float);
  float*    logfact2 = (float*)p;            p += 32 * sizeof(float);
  float*    priors2  = (float*)p;

  hmm_prep<<<1, 256, 0, stream>>>(lambdas, logA, priors, Tlin, loglam2, lamsum2,
                                  logfact2, priors2);
  hmm_chunk_scan<<<NCHUNK, 256, 0, stream>>>(x, Tlin, loglam2, lamsum2, logfact2, Mc);
  hmm_group_compose<<<NGROUP, 256, 0, stream>>>(Mc, Msuper);
  hmm_finalize<<<1, 256, 0, stream>>>(x, loglam2, lamsum2, logfact2, priors2, Msuper, out);
}
